// groupConvKAN_56152402428559
// MI455X (gfx1250) — matrix operand order repaired and checked by reading, not compiled, not run on hardware
//
#include <hip/hip_runtime.h>
#include <hip/hip_bf16.h>

typedef __attribute__((ext_vector_type(16))) _Float16 v16h;
typedef __attribute__((ext_vector_type(8)))  float    v8f;

#define BATCH 4
#define C_IN  128
#define HH    56
#define WW    56
#define GRP   4
#define CG    32          // channels per group
#define DPATCH 288        // CG * 9
#define NGRID 8
#define OG    32
#define LPIX  (HH * WW)   // 3136
#define KSPLINE 2304      // DPATCH * NGRID
#define KTOT   2592       // KSPLINE + DPATCH
#define TILES_PER_BG 196  // LPIX / 16
#define ROWSTRIDE 289     // 288 + 1 pad (odd stride -> conflict-free across rows)

// ---- prologue: convert & concatenate weights to f16: wcat[(g*OG+o)*2592 + f] ----
__global__ void kan_convert_weights(const float* __restrict__ spline_w,
                                    const float* __restrict__ base_w,
                                    _Float16* __restrict__ wcat, int n) {
    int i = blockIdx.x * blockDim.x + threadIdx.x;
    if (i >= n) return;
    int go = i / KTOT;
    int f  = i - go * KTOT;
    float v = (f < KSPLINE) ? spline_w[go * KSPLINE + f]
                            : base_w[go * DPATCH + (f - KSPLINE)];
    wcat[i] = (_Float16)v;
}

// ---- main fused kernel: im2col + RBF/SiLU feature gen + WMMA GEMM ----
__global__ void __launch_bounds__(64)
kan_wmma_kernel(const float* __restrict__ x,
                const _Float16* __restrict__ wcat,
                const float* __restrict__ base_b,
                float* __restrict__ out) {
    __shared__ float sP[2][16 * ROWSTRIDE];   // per-wave 16x288 fp32 patch tile

    const int lane   = threadIdx.x & 31;
    const int wvblk  = threadIdx.x >> 5;
    const int wgid   = blockIdx.x * 2 + wvblk;        // global wave id, [0, 3136)
    const int tile   = wgid % TILES_PER_BG;           // 16-pixel tile within (b,g)
    const int bg     = wgid / TILES_PER_BG;
    const int g      = bg & (GRP - 1);
    const int b      = bg >> 2;
    const int l0     = tile * 16;

    float* sp = &sP[wvblk][0];

    // ---- stage patch tile: p[row][d], d = c*9 + kh*3 + kw, zero-padded 3x3 ----
    const float* xin = x + ((size_t)b * C_IN + (size_t)g * CG) * (size_t)LPIX;
    for (int rr = 0; rr < 16; ++rr) {
        int l  = l0 + rr;
        int ho = l / WW;
        int wo = l - ho * WW;
        for (int d = lane; d < DPATCH; d += 32) {
            int c  = d / 9;
            int kq = d - c * 9;
            int kh = kq / 3;
            int kw = kq - kh * 3;
            int y  = ho - 1 + kh;
            int xw = wo - 1 + kw;
            float v = 0.0f;
            if ((unsigned)y < (unsigned)HH && (unsigned)xw < (unsigned)WW)
                v = xin[(size_t)c * LPIX + y * WW + xw];
            sp[rr * ROWSTRIDE + d] = v;
        }
    }
    __syncthreads();

    // ---- WMMA K-loop ----
    const int row = lane & 15;             // M index this lane supplies for A
    const int hi  = lane >> 4;             // lane half selects K sub-run
    const float* prow = sp + row * ROWSTRIDE;

    v8f acc0 = {};
    v8f acc1 = {};

    // B fragments: lane holds column o = lane&15, contiguous K run of 16 halves
    const _Float16* w0 = wcat + ((size_t)(g * OG) + (lane & 15)) * KTOT;
    const _Float16* w1 = w0 + (size_t)16 * KTOT;

    // spline part: 72 chunks of 32; chunk kk covers d in [kk*4, kk*4+4)
    for (int kk = 0; kk < 72; ++kk) {
        const int f0 = kk * 32 + hi * 16;
        v16h b0 = *(const v16h*)(w0 + f0);
        v16h b1 = *(const v16h*)(w1 + f0);

        const int d1 = kk * 4 + hi * 2;    // run1: K=hi*16..+7  -> one patch value
        const float p1 = prow[d1];
        const float p2 = prow[d1 + 1];     // run2: K=hi*16+8..+15

        v16h a;
#pragma unroll
        for (int t = 0; t < 8; ++t) {
            const float ct = -1.0f + (float)t * (2.0f / 7.0f);
            float u1 = (p1 - ct) * 3.5f;   // /h, h = 2/7
            float u2 = (p2 - ct) * 3.5f;
            a[t]     = (_Float16)__expf(-u1 * u1);
            a[t + 8] = (_Float16)__expf(-u2 * u2);
        }
        acc0 = __builtin_amdgcn_wmma_f32_16x16x32_f16(false, a, false, b0,
                                                      (short)0, acc0, false, false);
        acc1 = __builtin_amdgcn_wmma_f32_16x16x32_f16(false, a, false, b1,
                                                      (short)0, acc1, false, false);
    }

    // base part: 9 chunks of 32; features are silu(p_d)
    for (int kk = 72; kk < 81; ++kk) {
        const int f0 = kk * 32 + hi * 16;
        v16h b0 = *(const v16h*)(w0 + f0);
        v16h b1 = *(const v16h*)(w1 + f0);

        const int dbase = (kk - 72) * 32 + hi * 16;
        v16h a;
#pragma unroll
        for (int j = 0; j < 8; ++j) {
            float pa = prow[dbase + j];
            float pb = prow[dbase + 8 + j];
            a[j]     = (_Float16)(pa / (1.0f + __expf(-pa)));
            a[j + 8] = (_Float16)(pb / (1.0f + __expf(-pb)));
        }
        acc0 = __builtin_amdgcn_wmma_f32_16x16x32_f16(false, a, false, b0,
                                                      (short)0, acc0, false, false);
        acc1 = __builtin_amdgcn_wmma_f32_16x16x32_f16(false, a, false, b1,
                                                      (short)0, acc1, false, false);
    }

    // ---- epilogue: bias + scatter, group-major channel order ----
    const int n = lane & 15;
    const float bias0 = base_b[g * OG + n];
    const float bias1 = base_b[g * OG + 16 + n];
    float* outp = out + ((size_t)b * C_IN + (size_t)g * OG) * (size_t)LPIX;
#pragma unroll
    for (int r = 0; r < 8; ++r) {
        const int m = r + hi * 8;          // C/D layout: lanes 16-31 hold M=8..15
        const int l = l0 + m;
        outp[(size_t)n * LPIX + l]        = acc0[r] + bias0;
        outp[(size_t)(n + 16) * LPIX + l] = acc1[r] + bias1;
    }
}

extern "C" void kernel_launch(void* const* d_in, const int* in_sizes, int n_in,
                              void* d_out, int out_size, void* d_ws, size_t ws_size,
                              hipStream_t stream) {
    const float* x        = (const float*)d_in[0];
    const float* spline_w = (const float*)d_in[1];
    const float* base_w   = (const float*)d_in[2];
    const float* base_b   = (const float*)d_in[3];
    float* out            = (float*)d_out;

    _Float16* wcat = (_Float16*)d_ws;           // G*OG*2592 f16 = 663,552 bytes
    const int nconv = GRP * OG * KTOT;

    kan_convert_weights<<<(nconv + 255) / 256, 256, 0, stream>>>(
        spline_w, base_w, wcat, nconv);

    const int nwaves = BATCH * GRP * TILES_PER_BG;  // 3136 waves, 2 per block
    kan_wmma_kernel<<<nwaves / 2, 64, 0, stream>>>(x, wcat, base_b, out);
}
